// AudioQuantizer_18580028523005
// MI455X (gfx1250) — compile-verified, matrix operand order checked by reading, not run on hardware
//
#include <hip/hip_runtime.h>

typedef __attribute__((ext_vector_type(16))) _Float16 v16h;
typedef __attribute__((ext_vector_type(8)))  _Float16 v8h;
typedef __attribute__((ext_vector_type(8)))  float    v8f;
typedef __attribute__((ext_vector_type(4)))  float    v4f;

#define D_MODEL   512
#define NUM_CODES 1024
#define N_ROWS    65536
#define NT_BLOCK  8   // N-tiles (of 16 codes) accumulated simultaneously per wave

// ---------------------------------------------------------------------------
// Prep: split codebook f32 -> (hi,lo) f16 planes, compute exact ||c||^2.
// 1024 blocks (one per code) x 256 threads.
// ---------------------------------------------------------------------------
__global__ void vq_prep_kernel(const float* __restrict__ cb,
                               float* __restrict__ c_sq,
                               _Float16* __restrict__ cb_hi,
                               _Float16* __restrict__ cb_lo) {
    __shared__ float red[256];
    const int code = blockIdx.x;
    const int t    = threadIdx.x;
    float ssq = 0.0f;
#pragma unroll
    for (int i = 0; i < 2; ++i) {
        const int e = t + i * 256;
        const float v = cb[(size_t)code * D_MODEL + e];
        const _Float16 h = (_Float16)v;
        const float rem = v - (float)h;
        cb_hi[(size_t)code * D_MODEL + e] = h;
        cb_lo[(size_t)code * D_MODEL + e] = (_Float16)rem;
        ssq += v * v;
    }
    red[t] = ssq;
    __syncthreads();
    for (int s = 128; s > 0; s >>= 1) {
        if (t < s) red[t] += red[t + s];
        __syncthreads();
    }
    if (t == 0) c_sq[code] = red[0];
}

// ---------------------------------------------------------------------------
// Fragment loaders. WMMA 16-bit A (16x32) layout, lane l:
//   row = l & 15, kb = (l>>4)*8
//   frag[0..7]  <- K = k0 + kb + e
//   frag[8..15] <- K = k0 + 16 + kb + e
// B (32x16) is the mirror with col = l & 15; codebook row-major (code, d)
// makes both loads contiguous in K for a fixed row/col.
// ---------------------------------------------------------------------------
__device__ __forceinline__ void load_a_frag_split(const float* __restrict__ p,
                                                  v16h& hi, v16h& lo) {
    const v4f a0 = *(const v4f*)(p);
    const v4f a1 = *(const v4f*)(p + 4);
    const v4f a2 = *(const v4f*)(p + 16);
    const v4f a3 = *(const v4f*)(p + 20);
    const float vals[16] = {a0.x, a0.y, a0.z, a0.w, a1.x, a1.y, a1.z, a1.w,
                            a2.x, a2.y, a2.z, a2.w, a3.x, a3.y, a3.z, a3.w};
#pragma unroll
    for (int i = 0; i < 16; ++i) {
        const _Float16 h = (_Float16)vals[i];
        hi[i] = h;
        lo[i] = (_Float16)(vals[i] - (float)h);
    }
}

__device__ __forceinline__ v16h load_b_frag(const _Float16* __restrict__ p) {
    const v8h b0 = *(const v8h*)(p);
    const v8h b1 = *(const v8h*)(p + 16);
    v16h b;
#pragma unroll
    for (int i = 0; i < 8; ++i) { b[i] = b0[i]; b[8 + i] = b1[i]; }
    return b;
}

// ---------------------------------------------------------------------------
// Main: 4 waves / block, each wave handles one 16-row tile against all 1024
// codes with split-f16 WMMA (f32-accurate cross term), fused argmin + gather.
// ---------------------------------------------------------------------------
__global__ void __launch_bounds__(128) vq_main_kernel(
    const float* __restrict__ x,
    const float* __restrict__ c_sq,
    const _Float16* __restrict__ cb_hi,
    const _Float16* __restrict__ cb_lo,
    const float* __restrict__ emb,
    float* __restrict__ out) {

    const int lane = threadIdx.x & 31;
    const int wave = threadIdx.x >> 5;
    const int m0   = (blockIdx.x * 4 + wave) * 16;
    const int r    = lane & 15;          // row (A) / col (B) within tile
    const int kb   = (lane >> 4) * 8;    // K sub-chunk select

    const float* xrow = x + (size_t)(m0 + r) * D_MODEL;

    float best_score[8];
    int   best_idx[8];
#pragma unroll
    for (int j = 0; j < 8; ++j) { best_score[j] = 3.4e38f; best_idx[j] = 0; }

    for (int g = 0; g < NUM_CODES / (16 * NT_BLOCK); ++g) {   // 8 groups
        v8f acc[NT_BLOCK];
#pragma unroll
        for (int t = 0; t < NT_BLOCK; ++t) { v8f z = {}; acc[t] = z; }

        for (int ks = 0; ks < D_MODEL / 32; ++ks) {           // 16 k-steps
            const int k0 = ks * 32;
            v16h ahi, alo;
            load_a_frag_split(xrow + k0 + kb, ahi, alo);
#pragma unroll
            for (int t = 0; t < NT_BLOCK; ++t) {
                const int n = (g * NT_BLOCK + t) * 16 + r;
                const size_t boff = (size_t)n * D_MODEL + k0 + kb;
                const v16h bhi = load_b_frag(cb_hi + boff);
                const v16h blo = load_b_frag(cb_lo + boff);
                acc[t] = __builtin_amdgcn_wmma_f32_16x16x32_f16(
                    false, ahi, false, bhi, (short)0, acc[t], false, false);
                acc[t] = __builtin_amdgcn_wmma_f32_16x16x32_f16(
                    false, ahi, false, blo, (short)0, acc[t], false, false);
                acc[t] = __builtin_amdgcn_wmma_f32_16x16x32_f16(
                    false, alo, false, bhi, (short)0, acc[t], false, false);
            }
        }

        // score = ||c||^2 - 2 * (x . c); row m = j + 8*(lane>>4), col n
#pragma unroll
        for (int t = 0; t < NT_BLOCK; ++t) {
            const int n = (g * NT_BLOCK + t) * 16 + r;
            const float cs = c_sq[n];
#pragma unroll
            for (int j = 0; j < 8; ++j) {
                const float s = cs - 2.0f * acc[t][j];
                if (s < best_score[j]) { best_score[j] = s; best_idx[j] = n; }
            }
        }
    }

    // min-reduce across the 16 lanes of each half (same row m per j),
    // tie-break on the lower code index (argmin-first semantics).
#pragma unroll
    for (int j = 0; j < 8; ++j) {
#pragma unroll
        for (int m = 1; m < 16; m <<= 1) {
            const float os = __shfl_xor(best_score[j], m, 32);
            const int   oi = __shfl_xor(best_idx[j],   m, 32);
            if (os < best_score[j] ||
                (os == best_score[j] && oi < best_idx[j])) {
                best_score[j] = os;
                best_idx[j]   = oi;
            }
        }
    }

    // broadcast winner index for all 16 rows (rows 0-7 live in lane 0,
    // rows 8-15 in lane 16)
    int row_idx[16];
#pragma unroll
    for (int m = 0; m < 16; ++m)
        row_idx[m] = __shfl(best_idx[m & 7], (m >> 3) * 16, 32);

    // coalesced gather of the selected embedding rows
#pragma unroll
    for (int m = 0; m < 16; ++m) {
        const float* src = emb + (size_t)row_idx[m] * D_MODEL;
        float*       dst = out + (size_t)(m0 + m) * D_MODEL;
#pragma unroll
        for (int i = 0; i < 4; ++i) {
            *(v4f*)(dst + i * 128 + lane * 4) =
                *(const v4f*)(src + i * 128 + lane * 4);
        }
    }
}

// ---------------------------------------------------------------------------
extern "C" void kernel_launch(void* const* d_in, const int* in_sizes, int n_in,
                              void* d_out, int out_size, void* d_ws, size_t ws_size,
                              hipStream_t stream) {
    const float* x   = (const float*)d_in[0];   // (65536, 512) f32
    const float* cb  = (const float*)d_in[1];   // (1024, 512)  f32
    const float* emb = (const float*)d_in[2];   // (1024, 512)  f32
    float* out = (float*)d_out;                 // (65536, 512) f32

    char* ws = (char*)d_ws;
    float*    c_sq  = (float*)ws;                                       // 4 KB
    _Float16* cb_hi = (_Float16*)(ws + 4096);                           // 1 MB
    _Float16* cb_lo = (_Float16*)(ws + 4096 +
                                  (size_t)NUM_CODES * D_MODEL * sizeof(_Float16));

    vq_prep_kernel<<<NUM_CODES, 256, 0, stream>>>(cb, c_sq, cb_hi, cb_lo);
    vq_main_kernel<<<N_ROWS / 64, 128, 0, stream>>>(x, c_sq, cb_hi, cb_lo, emb, out);
}
